// RNNClassifier_46342697124486
// MI455X (gfx1250) — compile-verified, matrix operand order checked by reading, not run on hardware
//
#include <hip/hip_runtime.h>

#define V_ 50257
#define E_ 512
#define H_ 1024
#define S_ 4096

#define RNN_WGS 64
#define RNN_ROWS 16            /* H_/RNN_WGS */
#define W_PITCH (H_ + 16)      /* LDS row pitch: stride%64=16 -> no bank conflict */

typedef __attribute__((ext_vector_type(16))) __bf16 v16bf;
typedef __attribute__((ext_vector_type(8)))  float  v8f;
typedef __attribute__((ext_vector_type(4)))  int    v4i;

#define AS1 __attribute__((address_space(1)))
#define AS3 __attribute__((address_space(3)))

// ---------------------------------------------------------------- helpers ---
__device__ __forceinline__ unsigned short f2bf(float f) {
  unsigned u = __float_as_uint(f);
  u += 0x7FFFu + ((u >> 16) & 1u);     // round-to-nearest-even
  return (unsigned short)(u >> 16);
}

// -------------------------------------------------- phase 0: bf16 staging ---
__global__ void gather_embed_bf16(const int* __restrict__ tokens,
                                  const float* __restrict__ emb,
                                  unsigned short* __restrict__ xe) {
  int i = blockIdx.x * blockDim.x + threadIdx.x;   // [0, S*E)
  int s = i >> 9;                                  // /E_ (E_=512)
  int e = i & (E_ - 1);
  int tok = tokens[s];
  xe[i] = f2bf(emb[(long long)tok * E_ + e]);
}

__global__ void convert_w_bf16(const float* __restrict__ w,
                               unsigned short* __restrict__ wb, int n) {
  int i = blockIdx.x * blockDim.x + threadIdx.x;
  if (i < n) wb[i] = f2bf(w[i]);
}

// ------------------------------------- phase 1: x_proj GEMM via bf16 WMMA ---
// 8 waves/block share one M-tile (A rows hit WGP$) and cover 8 consecutive
// N-tiles. K = E_ = 512 -> 16 x v_wmma_f32_16x16x32_bf16 per tile.
__global__ void xproj_wmma(const unsigned short* __restrict__ xe,
                           const unsigned short* __restrict__ winb,
                           const float* __restrict__ b_in,
                           float* __restrict__ xproj) {
  const int lane  = threadIdx.x & 31;
  const int wave  = threadIdx.x >> 5;
  const int tm    = blockIdx.x >> 3;                    // 256 M-tiles
  const int tn    = ((blockIdx.x & 7) << 3) + wave;     // 64 N-tiles
  const int laneN = lane & 15;
  const int kHalf = (lane >> 4) << 4;                   // 0 or 16

  const unsigned short* arow = xe   + ((tm << 4) + laneN) * E_ + kHalf;
  const unsigned short* brow = winb + ((tn << 4) + laneN) * E_ + kHalf;

  v8f c = {};
#pragma unroll
  for (int k = 0; k < E_; k += 32) {
    v16bf a = *(const v16bf*)(arow + k);
    v16bf b = *(const v16bf*)(brow + k);
    c = __builtin_amdgcn_wmma_f32_16x16x32_bf16(false, a, false, b,
                                                (short)0, c, false, false);
  }

  // C/D layout: VGPR i -> M=i (lanes 0-15) / M=8+i (lanes 16-31), N=lane%16
  const int   n     = (tn << 4) + laneN;
  const float bias  = b_in[n];
  const int   mbase = (tm << 4) + ((lane >> 4) << 3);
#pragma unroll
  for (int i = 0; i < 8; ++i)
    xproj[(mbase + i) * H_ + n] = c[i] + bias;
}

// ------------------------------------------------ per-launch state reset ----
__global__ void init_state(float* __restrict__ hbuf, unsigned* __restrict__ sy) {
  int i = blockIdx.x * blockDim.x + threadIdx.x;
  if (i < 2 * H_) hbuf[i] = 0.0f;   // h0 = 0 (both buffers)
  if (i < 2)      sy[i]   = 0u;     // barrier counter + generation
}

// ------------------- phase 2: persistent sequential scan, W_h pinned in LDS -
__global__ void rnn_scan(const float* __restrict__ W_h,
                         const float* __restrict__ b_h,
                         const float* __restrict__ xproj,
                         float* __restrict__ hbuf,
                         unsigned* __restrict__ sy) {
  __shared__ float wsh[RNN_ROWS * W_PITCH];    // ~65 KB, padded rows
  __shared__ float hsh[H_];                    // 4 KB staged h_t
  const int tid  = threadIdx.x;                // 256 threads = 8 waves
  const int row0 = blockIdx.x * RNN_ROWS;

  // ---- stage this WG's W_h rows into LDS (async DMA path if available) ----
  for (int rr = 0; rr < RNN_ROWS; ++rr) {
    const float* src = W_h + (row0 + rr) * H_;
    float*       dst = &wsh[rr * W_PITCH];
#if __has_builtin(__builtin_amdgcn_global_load_async_to_lds_b128)
    for (int i = tid * 4; i < H_; i += 256 * 4)
      __builtin_amdgcn_global_load_async_to_lds_b128(
          (AS1 v4i*)(src + i), (AS3 v4i*)(dst + i), 0, 0);
#else
    for (int i = tid; i < H_; i += 256) dst[i] = src[i];
#endif
  }
#if __has_builtin(__builtin_amdgcn_global_load_async_to_lds_b128)
#if __has_builtin(__builtin_amdgcn_s_wait_asynccnt)
  __builtin_amdgcn_s_wait_asynccnt(0);
#else
  asm volatile("s_wait_asynccnt 0x0" ::: "memory");
#endif
#endif
  __syncthreads();

  const int   r     = tid >> 4;                // row within WG (0..15)
  const int   cpart = tid & 15;                // 16 lanes cooperate per row
  const int   grow  = row0 + r;
  const float bh    = b_h[grow];
  const float* wrow = &wsh[r * W_PITCH];

  unsigned* cnt = &sy[0];
  unsigned* gen = &sy[1];

  for (int t = 0; t < S_; ++t) {
    const float* hcur = hbuf + ((t & 1) << 10);
    float*       hnxt = hbuf + (((t + 1) & 1) << 10);
    if (t + 1 < S_)
      __builtin_prefetch(&xproj[(t + 1) * H_ + grow], 0, 1); // global_prefetch_b8

    // stage h_t into LDS: 4 device-scope loads per thread (bypass WGP$)
    {
      const int i = tid << 2;
      float a0 = __hip_atomic_load(&hcur[i + 0], __ATOMIC_RELAXED, __HIP_MEMORY_SCOPE_AGENT);
      float a1 = __hip_atomic_load(&hcur[i + 1], __ATOMIC_RELAXED, __HIP_MEMORY_SCOPE_AGENT);
      float a2 = __hip_atomic_load(&hcur[i + 2], __ATOMIC_RELAXED, __HIP_MEMORY_SCOPE_AGENT);
      float a3 = __hip_atomic_load(&hcur[i + 3], __ATOMIC_RELAXED, __HIP_MEMORY_SCOPE_AGENT);
      hsh[i + 0] = a0; hsh[i + 1] = a1; hsh[i + 2] = a2; hsh[i + 3] = a3;
    }
    __syncthreads();

    // interleaved dot: lane cpart handles columns cpart, cpart+16, ...
    // -> consecutive lanes touch consecutive LDS words (conflict-free)
    float sum = 0.0f;
#pragma unroll 8
    for (int jj = 0; jj < 64; ++jj) {
      const int idx = cpart + (jj << 4);
      sum = __builtin_fmaf(wrow[idx], hsh[idx], sum);
    }
    // reduce across the 16-lane group (wave32-safe ds_swizzle SWAPX patterns)
    sum += __int_as_float(__builtin_amdgcn_ds_swizzle(__float_as_int(sum), 0x041F));
    sum += __int_as_float(__builtin_amdgcn_ds_swizzle(__float_as_int(sum), 0x081F));
    sum += __int_as_float(__builtin_amdgcn_ds_swizzle(__float_as_int(sum), 0x101F));
    sum += __int_as_float(__builtin_amdgcn_ds_swizzle(__float_as_int(sum), 0x201F));

    if (cpart == 0) {
      float y = xproj[t * H_ + grow] + bh + sum;
      __hip_atomic_store(&hnxt[grow], tanhf(y), __ATOMIC_RELAXED,
                         __HIP_MEMORY_SCOPE_AGENT);
    }

    // -------- device-wide barrier: counter + generation, AGENT acq/rel -----
    __syncthreads();
    if (tid == 0) {
      unsigned old = __hip_atomic_fetch_add(cnt, 1u, __ATOMIC_ACQ_REL,
                                            __HIP_MEMORY_SCOPE_AGENT);
      if (old == RNN_WGS - 1) {
        __hip_atomic_store(cnt, 0u, __ATOMIC_RELAXED, __HIP_MEMORY_SCOPE_AGENT);
        __hip_atomic_store(gen, (unsigned)(t + 1), __ATOMIC_RELEASE,
                           __HIP_MEMORY_SCOPE_AGENT);
      } else {
        while (__hip_atomic_load(gen, __ATOMIC_ACQUIRE,
                                 __HIP_MEMORY_SCOPE_AGENT) < (unsigned)(t + 1))
          __builtin_amdgcn_s_sleep(1);
      }
    }
    __syncthreads();
  }
}

// --------------------------------------- phase 3: head + log_softmax --------
__global__ void head_logsoftmax(const float* __restrict__ hfin,
                                const float* __restrict__ W_out,
                                const float* __restrict__ b_out,
                                float* __restrict__ out) {
  __shared__ float s[2];
  const int lane = threadIdx.x & 31;
  const int w    = threadIdx.x >> 5;           // class 0 / 1 (64 threads)
  float sum = 0.0f;
#pragma unroll
  for (int j = 0; j < H_ / 32; ++j)
    sum = __builtin_fmaf(W_out[w * H_ + j * 32 + lane], hfin[j * 32 + lane], sum);
  sum += __int_as_float(__builtin_amdgcn_ds_swizzle(__float_as_int(sum), 0x041F));
  sum += __int_as_float(__builtin_amdgcn_ds_swizzle(__float_as_int(sum), 0x081F));
  sum += __int_as_float(__builtin_amdgcn_ds_swizzle(__float_as_int(sum), 0x101F));
  sum += __int_as_float(__builtin_amdgcn_ds_swizzle(__float_as_int(sum), 0x201F));
  sum += __int_as_float(__builtin_amdgcn_ds_swizzle(__float_as_int(sum), 0x401F));
  if (lane == 0) s[w] = sum + b_out[w];
  __syncthreads();
  if (threadIdx.x == 0) {
    float s0 = s[0], s1 = s[1];
    float m  = fmaxf(s0, s1);
    float l  = logf(__expf(s0 - m) + __expf(s1 - m));
    out[0] = s0 - m - l;
    out[1] = s1 - m - l;
  }
}

// ---------------------------------------------------------------- launch ----
extern "C" void kernel_launch(void* const* d_in, const int* in_sizes, int n_in,
                              void* d_out, int out_size, void* d_ws, size_t ws_size,
                              hipStream_t stream) {
  const int*   tokens = (const int*)  d_in[0];
  const float* emb    = (const float*)d_in[1];
  const float* W_in   = (const float*)d_in[2];
  const float* b_in   = (const float*)d_in[3];
  const float* W_h    = (const float*)d_in[4];
  const float* b_h    = (const float*)d_in[5];
  const float* W_out  = (const float*)d_in[6];
  const float* b_out  = (const float*)d_in[7];
  float* out = (float*)d_out;

  char* ws = (char*)d_ws;
  size_t off = 0;
  unsigned short* xe = (unsigned short*)(ws + off); off += (size_t)S_ * E_ * 2; // 4 MB
  unsigned short* wb = (unsigned short*)(ws + off); off += (size_t)H_ * E_ * 2; // 1 MB
  float*    xproj = (float*)(ws + off);    off += (size_t)S_ * H_ * 4;          // 16 MB
  float*    hbuf  = (float*)(ws + off);    off += (size_t)2 * H_ * 4;           // 8 KB
  unsigned* syncb = (unsigned*)(ws + off); off += 256;

  gather_embed_bf16<<<(S_ * E_) / 256, 256, 0, stream>>>(tokens, emb, xe);
  convert_w_bf16<<<(H_ * E_) / 256, 256, 0, stream>>>(W_in, wb, H_ * E_);
  xproj_wmma<<<(S_ / 16) * (H_ / 16) / 8, 256, 0, stream>>>(xe, wb, b_in, xproj);
  init_state<<<8, 256, 0, stream>>>(hbuf, syncb);
  rnn_scan<<<RNN_WGS, 256, 0, stream>>>(W_h, b_h, xproj, hbuf, syncb);
  head_logsoftmax<<<1, 64, 0, stream>>>(hbuf, W_out, b_out, out);
}
